// CausalSelfAttention_43207370998448
// MI455X (gfx1250) — matrix operand order repaired and checked by reading, not compiled, not run on hardware
//
#include <hip/hip_runtime.h>

// ---------------------------------------------------------------------------
// Causal self-attention forward for MI455X (gfx1250, wave32, WMMA).
//   B=4, T=2048, C=1024, H=16, D=64
// Pipeline: qkv_gemm (f32->f16 WMMA, double-buffered LDS)
//        -> flash_attn (WMMA + online softmax, async G->LDS copies)
//        -> proj_gemm (f16 A via async G->LDS, f32 out, double-buffered).
// All matrix math uses v_wmma_f32_16x16x32_f16 (f32 accumulate).
// Workspace (f16): Q | K | V (each B*H*T*D = 8388608 elems) | Y (B*T*C)
// ---------------------------------------------------------------------------

typedef _Float16 v16h  __attribute__((ext_vector_type(16)));
typedef _Float16 half8 __attribute__((ext_vector_type(8)));
typedef _Float16 half4 __attribute__((ext_vector_type(4)));
typedef float    v8f   __attribute__((ext_vector_type(8)));
typedef float    f32x4 __attribute__((ext_vector_type(4)));
typedef int      v4i   __attribute__((ext_vector_type(4)));

static constexpr int Bsz = 4, Tseq = 2048, Cdim = 1024, Hn = 16, Dh = 64;
static constexpr int Mrows = Bsz * Tseq;          // 8192
static constexpr float SCALE = 0.125f;            // 1/sqrt(64)
static constexpr float NEGINF_ADD = -1.0e9f;

// --------- CDNA5 async global->LDS copy (ASYNCcnt path), with fallback ------
#if defined(__has_builtin)
#if __has_builtin(__builtin_amdgcn_global_load_async_to_lds_b128)
#define HAVE_ASYNC_LDS 1
#endif
#endif

__device__ __forceinline__ void copy16_g2l(_Float16* l, const _Float16* g) {
#if defined(HAVE_ASYNC_LDS)
    // signature (probe-derived): (global int4* src, local int4* dst, imm off, imm cpol)
    __builtin_amdgcn_global_load_async_to_lds_b128(
        (__attribute__((address_space(1))) v4i*)(void*)g,
        (__attribute__((address_space(3))) v4i*)(void*)l, 0, 0);
#else
    *(half8*)l = *(const half8*)g;
#endif
}

__device__ __forceinline__ void async_wait0() {
#if defined(HAVE_ASYNC_LDS)
#if __has_builtin(__builtin_amdgcn_s_wait_asynccnt)
    __builtin_amdgcn_s_wait_asynccnt(0);
#else
    asm volatile("s_wait_asynccnt 0" ::: "memory");
#endif
#endif
}

// Concatenate two 8-half LDS chunks into one 16-half WMMA fragment.
__device__ __forceinline__ v16h frag_cat(const _Float16* p0, const _Float16* p1) {
    half8 lo = *(const half8*)p0;
    half8 hi = *(const half8*)p1;
    return __builtin_shufflevector(lo, hi, 0, 1, 2, 3, 4, 5, 6, 7,
                                           8, 9, 10, 11, 12, 13, 14, 15);
}

__device__ __forceinline__ v8f wmma_f16(v16h a, v16h b, v8f c) {
    return __builtin_amdgcn_wmma_f32_16x16x32_f16(
        false, a, false, b, (short)0, c, false, false);
}

// ---------------------------------------------------------------------------
// Kernel 1: qkv = x @ W_attn + b_attn ; scatter f16 into head-major Q,K,V.
// Grid: (64, 24). Block: 256 = 8 waves. LDS double-buffered.
// ---------------------------------------------------------------------------
__global__ __launch_bounds__(256) void qkv_gemm(
    const float* __restrict__ X,      // [8192,1024]
    const float* __restrict__ W,      // [1024,3072]
    const float* __restrict__ bias,   // [3072]
    _Float16* __restrict__ Qo, _Float16* __restrict__ Ko, _Float16* __restrict__ Vo) {

    __shared__ _Float16 As[2][128 * 32];   // [m][k]
    __shared__ _Float16 Bt[2][128 * 32];   // [n][k] (transposed)

    const int tid  = threadIdx.x;
    const int lane = tid & 31;
    const int wave = tid >> 5;
    const int wm   = (wave >> 2) * 64;
    const int wn   = (wave & 3) * 32;
    const int mbase = blockIdx.x * 128;
    const int nbase = blockIdx.y * 128;

    const int row16  = lane & 15;
    const int koff8  = (lane < 16) ? 0 : 8;
    const int koff16 = (lane < 16) ? 0 : 16;
    const int rbase  = (lane < 16) ? 0 : 8;

    const v8f vzero = {0.f, 0.f, 0.f, 0.f, 0.f, 0.f, 0.f, 0.f};
    v8f acc[4][2];
    #pragma unroll
    for (int i = 0; i < 4; ++i)
        #pragma unroll
        for (int j = 0; j < 2; ++j) acc[i][j] = vzero;

    f32x4 ax[4], bx[4];   // register-staged next tile

    auto load_regs = [&](int kt) {
        #pragma unroll
        for (int it = 0; it < 4; ++it) {
            int idx = (tid + it * 256) * 4;
            int ra = idx >> 5, ca = idx & 31;
            ax[it] = *(const f32x4*)&X[(size_t)(mbase + ra) * Cdim + kt + ca];
            int rb = idx >> 7, cb = idx & 127;
            bx[it] = *(const f32x4*)&W[(size_t)(kt + rb) * (3 * Cdim) + nbase + cb];
        }
    };
    auto store_lds = [&](int buf) {
        #pragma unroll
        for (int it = 0; it < 4; ++it) {
            int idx = (tid + it * 256) * 4;
            int ra = idx >> 5, ca = idx & 31;
            half4 ha;
            #pragma unroll
            for (int u = 0; u < 4; ++u) ha[u] = (_Float16)ax[it][u];
            *(half4*)&As[buf][ra * 32 + ca] = ha;
            int rb = idx >> 7, cb = idx & 127;
            #pragma unroll
            for (int u = 0; u < 4; ++u)
                Bt[buf][(cb + u) * 32 + rb] = (_Float16)bx[it][u];
        }
    };

    load_regs(0);
    store_lds(0);
    __syncthreads();

    int buf = 0;
    for (int kt = 0; kt < Cdim; kt += 32) {
        const bool nxt = (kt + 32) < Cdim;
        if (nxt) load_regs(kt + 32);   // global loads overlap WMMA below

        v16h bfr[2];
        #pragma unroll
        for (int nt = 0; nt < 2; ++nt) {
            const _Float16* p = &Bt[buf][(wn + nt * 16 + row16) * 32 + koff16];
            bfr[nt] = frag_cat(p, p + 8);
        }
        #pragma unroll
        for (int mt = 0; mt < 4; ++mt) {
            const _Float16* pa = &As[buf][(wm + mt * 16 + row16) * 32];
            v16h afr = frag_cat(pa + koff16, pa + koff16 + 8);
            #pragma unroll
            for (int nt = 0; nt < 2; ++nt)
                acc[mt][nt] = wmma_f16(afr, bfr[nt], acc[mt][nt]);
        }

        if (nxt) store_lds(buf ^ 1);   // fill other buffer; peers still read buf
        __syncthreads();
        buf ^= 1;
    }

    // epilogue: bias add, scatter into head-major Q/K/V f16
    #pragma unroll
    for (int mt = 0; mt < 4; ++mt) {
        #pragma unroll
        for (int nt = 0; nt < 2; ++nt) {
            int gcol = nbase + wn + nt * 16 + row16;       // [0, 3072)
            float bv = bias[gcol];
            int sel = gcol >> 10;                          // 0:Q 1:K 2:V
            int cc  = gcol & 1023;
            int h   = cc >> 6, d = cc & 63;
            _Float16* dst = (sel == 0) ? Qo : (sel == 1) ? Ko : Vo;
            #pragma unroll
            for (int r = 0; r < 8; ++r) {
                int grow = mbase + wm + mt * 16 + rbase + r;
                int b = grow >> 11, t = grow & 2047;
                dst[(((size_t)(b * Hn + h) * Tseq + t) * Dh) + d] =
                    (_Float16)(acc[mt][nt][r] + bv);
            }
        }
    }
}

// ---------------------------------------------------------------------------
// Kernel 2: flash attention, causal. Grid: (T/128, B*H) = (16, 64).
// ---------------------------------------------------------------------------
__global__ __launch_bounds__(256) void flash_attn(
    const _Float16* __restrict__ Q,   // [B,H,T,D]
    const _Float16* __restrict__ K,
    const _Float16* __restrict__ V,
    _Float16* __restrict__ Y) {       // [B,T,C]

    __shared__ _Float16 Qs[128 * 64];     // [qrow][d]
    __shared__ _Float16 Ks[64 * 64];      // [key][d]
    __shared__ _Float16 Vt[64 * 64];      // [d][key] (transposed)
    __shared__ _Float16 Ps[8][16 * 64];   // per-wave P tile [qrow][key]

    const int tid  = threadIdx.x;
    const int lane = tid & 31;
    const int wave = tid >> 5;
    const int qbase = blockIdx.x * 128;
    const int bh = blockIdx.y;
    const int b = bh >> 4, h = bh & 15;

    const _Float16* Qg = Q + ((size_t)bh * Tseq + qbase) * Dh;
    const _Float16* Kg = K + (size_t)bh * Tseq * Dh;
    const _Float16* Vg = V + (size_t)bh * Tseq * Dh;

    // stage the 128x64 Q tile once (async G->LDS)
    #pragma unroll
    for (int it = 0; it < 4; ++it) {
        int idx = (tid + it * 256) * 8;
        copy16_g2l(&Qs[idx], &Qg[idx]);
    }

    const int row16  = lane & 15;
    const int koff8  = (lane < 16) ? 0 : 8;
    const int koff16 = (lane < 16) ? 0 : 16;
    const int rbase  = (lane < 16) ? 0 : 8;

    const v8f vzero = {0.f, 0.f, 0.f, 0.f, 0.f, 0.f, 0.f, 0.f};
    v8f o[4];
    float m[8], l[8];
    #pragma unroll
    for (int dt = 0; dt < 4; ++dt) o[dt] = vzero;
    #pragma unroll
    for (int r = 0; r < 8; ++r) { m[r] = -__builtin_inff(); l[r] = 0.f; }

    const int nkb = 2 * blockIdx.x + 2;   // causal: key blocks <= q block
    for (int j = 0; j < nkb; ++j) {
        const int kb = j * 64;
        __syncthreads();   // protect Ks/Vt reuse from previous iteration

        // K tile: straight async copy; V tile: reg-staged then transposed store
        half8 vv[2];
        #pragma unroll
        for (int it = 0; it < 2; ++it) {
            int idx = (tid + it * 256) * 8;
            copy16_g2l(&Ks[idx], &Kg[(size_t)kb * Dh + idx]);
            vv[it] = *(const half8*)&Vg[(size_t)kb * Dh + idx];
        }
        #pragma unroll
        for (int it = 0; it < 2; ++it) {
            int idx = (tid + it * 256) * 8;
            int r = idx >> 6, c = idx & 63;
            #pragma unroll
            for (int u = 0; u < 8; ++u) Vt[(c + u) * 64 + r] = vv[it][u];
        }
        async_wait0();
        __syncthreads();

        // ---- S = Q K^T (16x64 per wave) ----
        v8f s[4];
        #pragma unroll
        for (int nt = 0; nt < 4; ++nt) s[nt] = vzero;
        #pragma unroll
        for (int kc = 0; kc < Dh; kc += 32) {
            const _Float16* pa = &Qs[(wave * 16 + row16) * 64 + kc];
            v16h afr = frag_cat(pa + koff16, pa + koff16 + 8);
            #pragma unroll
            for (int nt = 0; nt < 4; ++nt) {
                const _Float16* pb = &Ks[(nt * 16 + row16) * 64 + kc + koff16];
                s[nt] = wmma_f16(afr, frag_cat(pb, pb + 8), s[nt]);
            }
        }

        // ---- scale + causal additive mask + row max ----
        float rowmax[8];
        #pragma unroll
        for (int r = 0; r < 8; ++r) rowmax[r] = -__builtin_inff();
        #pragma unroll
        for (int nt = 0; nt < 4; ++nt) {
            int key = kb + nt * 16 + row16;
            #pragma unroll
            for (int r = 0; r < 8; ++r) {
                int qrow = qbase + wave * 16 + rbase + r;
                float sv = s[nt][r] * SCALE;
                if (key > qrow) sv += NEGINF_ADD;
                s[nt][r] = sv;
                rowmax[r] = fmaxf(rowmax[r], sv);
            }
        }
        #pragma unroll
        for (int msk = 1; msk < 16; msk <<= 1)
            #pragma unroll
            for (int r = 0; r < 8; ++r)
                rowmax[r] = fmaxf(rowmax[r], __shfl_xor(rowmax[r], msk, 32));

        // ---- online softmax rescale ----
        float alpha[8];
        #pragma unroll
        for (int r = 0; r < 8; ++r) {
            float mn = fmaxf(m[r], rowmax[r]);
            alpha[r] = __expf(m[r] - mn);
            m[r] = mn;
            l[r] *= alpha[r];
        }
        #pragma unroll
        for (int dt = 0; dt < 4; ++dt)
            #pragma unroll
            for (int r = 0; r < 8; ++r) o[dt][r] *= alpha[r];

        // ---- P = exp(S - m), row sums, stage P (A-operand layout) ----
        float rs[8];
        #pragma unroll
        for (int r = 0; r < 8; ++r) rs[r] = 0.f;
        #pragma unroll
        for (int nt = 0; nt < 4; ++nt)
            #pragma unroll
            for (int r = 0; r < 8; ++r) {
                float p = __expf(s[nt][r] - m[r]);
                rs[r] += p;
                Ps[wave][(rbase + r) * 64 + nt * 16 + row16] = (_Float16)p;
            }
        #pragma unroll
        for (int msk = 1; msk < 16; msk <<= 1)
            #pragma unroll
            for (int r = 0; r < 8; ++r) rs[r] += __shfl_xor(rs[r], msk, 32);
        #pragma unroll
        for (int r = 0; r < 8; ++r) l[r] += rs[r];

        // ---- O += P V ----
        #pragma unroll
        for (int kc = 0; kc < 64; kc += 32) {
            const _Float16* pa = &Ps[wave][row16 * 64 + kc];
            v16h afr = frag_cat(pa + koff16, pa + koff16 + 8);
            #pragma unroll
            for (int dt = 0; dt < 4; ++dt) {
                const _Float16* pb = &Vt[(dt * 16 + row16) * 64 + kc + koff16];
                o[dt] = wmma_f16(afr, frag_cat(pb, pb + 8), o[dt]);
            }
        }
    }

    // ---- normalize and write Y (f16, [B,T,C]) ----
    float inv[8];
    #pragma unroll
    for (int r = 0; r < 8; ++r) inv[r] = 1.0f / l[r];
    #pragma unroll
    for (int dt = 0; dt < 4; ++dt) {
        int d = dt * 16 + row16;
        #pragma unroll
        for (int r = 0; r < 8; ++r) {
            int qrow = qbase + wave * 16 + rbase + r;
            Y[((size_t)(b * Tseq + qrow) * Cdim) + h * Dh + d] =
                (_Float16)(o[dt][r] * inv[r]);
        }
    }
}

// ---------------------------------------------------------------------------
// Kernel 3: out = Y @ W_proj + b_proj. A is f16 (async G->LDS), out f32.
// Grid: (64, 8). Block: 256. LDS double-buffered.
// ---------------------------------------------------------------------------
__global__ __launch_bounds__(256) void proj_gemm(
    const _Float16* __restrict__ A,   // [8192,1024] f16
    const float* __restrict__ W,      // [1024,1024]
    const float* __restrict__ bias,   // [1024]
    float* __restrict__ Out) {        // [8192,1024]

    __shared__ _Float16 As[2][128 * 32];
    __shared__ _Float16 Bt[2][128 * 32];

    const int tid  = threadIdx.x;
    const int lane = tid & 31;
    const int wave = tid >> 5;
    const int wm   = (wave >> 2) * 64;
    const int wn   = (wave & 3) * 32;
    const int mbase = blockIdx.x * 128;
    const int nbase = blockIdx.y * 128;

    const int row16  = lane & 15;
    const int koff8  = (lane < 16) ? 0 : 8;
    const int koff16 = (lane < 16) ? 0 : 16;
    const int rbase  = (lane < 16) ? 0 : 8;

    const v8f vzero = {0.f, 0.f, 0.f, 0.f, 0.f, 0.f, 0.f, 0.f};
    v8f acc[4][2];
    #pragma unroll
    for (int i = 0; i < 4; ++i)
        #pragma unroll
        for (int j = 0; j < 2; ++j) acc[i][j] = vzero;

    f32x4 bx[4];

    auto stage_A = [&](int kt, int buf) {       // f16 tile: async G->LDS
        #pragma unroll
        for (int it = 0; it < 2; ++it) {
            int idx = (tid + it * 256) * 8;
            int r = idx >> 5, c = idx & 31;
            copy16_g2l(&As[buf][r * 32 + c],
                       &A[(size_t)(mbase + r) * Cdim + kt + c]);
        }
    };
    auto load_B = [&](int kt) {
        #pragma unroll
        for (int it = 0; it < 4; ++it) {
            int idx = (tid + it * 256) * 4;
            int r = idx >> 7, c = idx & 127;
            bx[it] = *(const f32x4*)&W[(size_t)(kt + r) * Cdim + nbase + c];
        }
    };
    auto store_B = [&](int buf) {
        #pragma unroll
        for (int it = 0; it < 4; ++it) {
            int idx = (tid + it * 256) * 4;
            int r = idx >> 7, c = idx & 127;
            #pragma unroll
            for (int u = 0; u < 4; ++u)
                Bt[buf][(c + u) * 32 + r] = (_Float16)bx[it][u];
        }
    };

    stage_A(0, 0);
    load_B(0);
    store_B(0);
    async_wait0();
    __syncthreads();

    int buf = 0;
    for (int kt = 0; kt < Cdim; kt += 32) {
        const bool nxt = (kt + 32) < Cdim;
        if (nxt) { stage_A(kt + 32, buf ^ 1); load_B(kt + 32); }

        v16h bfr[2];
        #pragma unroll
        for (int nt = 0; nt < 2; ++nt) {
            const _Float16* p = &Bt[buf][(wn + nt * 16 + row16) * 32 + koff16];
            bfr[nt] = frag_cat(p, p + 8);
        }
        #pragma unroll
        for (int mt = 0; mt < 4; ++mt) {
            const _Float16* pa = &As[buf][(wm + mt * 16 + row16) * 32];
            v16h afr = frag_cat(pa + koff16, pa + koff16 + 8);
            #pragma unroll
            for (int nt = 0; nt < 2; ++nt)
                acc[mt][nt] = wmma_f16(afr, bfr[nt], acc[mt][nt]);
        }

        if (nxt) store_B(buf ^ 1);
        async_wait0();
        __syncthreads();
        buf ^= 1;
    }

    #pragma unroll
    for (int mt = 0; mt < 4; ++mt)
        #pragma unroll
        for (int nt = 0; nt < 2; ++nt) {
            int gcol = nbase + wn + nt * 16 + row16;
            float bv = bias[gcol];
            #pragma unroll
            for (int r = 0; r < 8; ++r) {
                int grow = mbase + wm + mt * 16 + rbase + r;
                Out[(size_t)grow * Cdim + gcol] = acc[mt][nt][r] + bv;
            }
        }
}

// ---------------------------------------------------------------------------
extern "C" void kernel_launch(void* const* d_in, const int* in_sizes, int n_in,
                              void* d_out, int out_size, void* d_ws, size_t ws_size,
                              hipStream_t stream) {
    const float* x      = (const float*)d_in[0];
    // d_in[1] = mask: exactly tril(ones) -> causal masking applied analytically.
    const float* W_attn = (const float*)d_in[2];
    const float* b_attn = (const float*)d_in[3];
    const float* W_proj = (const float*)d_in[4];
    const float* b_proj = (const float*)d_in[5];
    float* out = (float*)d_out;

    const size_t QKV_ELEMS = (size_t)Bsz * Hn * Tseq * Dh;   // 8388608
    _Float16* Qh = (_Float16*)d_ws;
    _Float16* Kh = Qh + QKV_ELEMS;
    _Float16* Vh = Kh + QKV_ELEMS;
    _Float16* Yh = Vh + QKV_ELEMS;

    qkv_gemm<<<dim3(Mrows / 128, (3 * Cdim) / 128), 256, 0, stream>>>(
        x, W_attn, b_attn, Qh, Kh, Vh);
    flash_attn<<<dim3(Tseq / 128, Bsz * Hn), 256, 0, stream>>>(Qh, Kh, Vh, Yh);
    proj_gemm<<<dim3(Mrows / 128, Cdim / 128), 256, 0, stream>>>(
        Yh, W_proj, b_proj, out);
}
